// AttentionBlock_21388937134396
// MI455X (gfx1250) — matrix operand order repaired and checked by reading, not compiled, not run on hardware
//
#include <hip/hip_runtime.h>
#include <hip/hip_bf16.h>
#include <math.h>

// ---------------------------------------------------------------------------
// Types for CDNA5 WMMA (wave32, 16x16x32 bf16 -> f32)
// ---------------------------------------------------------------------------
typedef __bf16 bf16;
typedef __attribute__((ext_vector_type(2)))  __bf16 v2bf;
typedef __attribute__((ext_vector_type(8)))  __bf16 v8bf;
typedef __attribute__((ext_vector_type(16))) __bf16 v16bf;
typedef __attribute__((ext_vector_type(8)))  float  v8f;
typedef __attribute__((ext_vector_type(4)))  float  v4f;
typedef __attribute__((ext_vector_type(4)))  unsigned int v4u;
typedef __attribute__((ext_vector_type(4)))  int v4i;
typedef __attribute__((ext_vector_type(8)))  int v8i;

union V16U { v16bf v; v8bf h[2]; };

#define Bn 8
#define Cn 512
#define Sn 1024
#define NH 8
#define Dn 64

#define USE_TDM 1

// scale = 1/sqrt(64) * log2(e): softmax runs in exp2 domain
#define QSCALE 0.18033688f

__device__ __forceinline__ float fast_exp2(float x) {
  return __builtin_amdgcn_exp2f(x);
}

// truncating fp32->bf16: free (folds into *_store_d16_hi)
__device__ __forceinline__ bf16 f2bf_trunc(float f) {
  unsigned u = __builtin_bit_cast(unsigned, f);
  return __builtin_bit_cast(bf16, (unsigned short)(u >> 16));
}

// pack two fp32 -> packed bf16 pair. HW cvt if present, else 1x v_perm_b32.
__device__ __forceinline__ v2bf pack_bf(float a, float b) {
#if __has_builtin(__builtin_amdgcn_cvt_pk_bf16_f32)
  return __builtin_amdgcn_cvt_pk_bf16_f32(a, b);
#else
  unsigned ua = __builtin_bit_cast(unsigned, a);
  unsigned ub = __builtin_bit_cast(unsigned, b);
  return __builtin_bit_cast(v2bf, __builtin_amdgcn_perm(ub, ua, 0x07060302u));
#endif
}

__device__ __forceinline__ v8f wmma_bf16(v16bf a, v16bf b, v8f c) {
  return __builtin_amdgcn_wmma_f32_16x16x32_bf16(false, a, false, b,
                                                 (short)0, c, false, false);
}

// A fragment (16xK=32): lane m = lane&15, khalf = lane>>4.
__device__ __forceinline__ v16bf load_a_frag(const bf16* rowbase, int khalf) {
  V16U u;
  u.h[0] = *(const v8bf*)(rowbase + khalf * 8);
  u.h[1] = *(const v8bf*)(rowbase + 16 + khalf * 8);
  return u.v;
}
// B fragment (K=32 x 16): lane n = lane&15, K = (lane>>4)*8 + [0..8) and 16 + (lane>>4)*8 + [0..8)
__device__ __forceinline__ v16bf load_b_frag(const bf16* colbase, int lanehi) {
  V16U u;
  const bf16* p = colbase + lanehi * 8;
  u.h[0] = *(const v8bf*)(p);
  u.h[1] = *(const v8bf*)(p + 16);
  return u.v;
}

// CDNA5 async global->LDS copy (ASYNCcnt). Per-lane 16B: LDS[vdst]=MEM[vaddr].
__device__ __forceinline__ void async_ld_b128(unsigned lds_off,
                                              unsigned long long gaddr) {
  asm volatile("global_load_async_to_lds_b128 %0, %1, off"
               :: "v"(lds_off), "v"(gaddr) : "memory");
}
__device__ __forceinline__ void wait_async0() {
  asm volatile("s_wait_asynccnt 0x0" ::: "memory");
}
// stage 32 contiguous bf16 (64B) via 4 async b128 ops
__device__ __forceinline__ void stage32_async(bf16* ldst, const bf16* gsrc) {
#pragma unroll
  for (int i = 0; i < 4; i++)
    async_ld_b128((unsigned)(size_t)(ldst + 8 * i),
                  (unsigned long long)(size_t)(gsrc + 8 * i));
}

#if USE_TDM && __has_builtin(__builtin_amdgcn_tensor_load_to_lds)
#define HAVE_TDM 1
// 2D TENSOR_LOAD_TO_LDS: tile tw x th bf16, row stride `stride` (elements),
// contiguous LDS dest. D# bitfields per CDNA5 ISA 8.3/8.4.
__device__ __forceinline__ void tdm_load_2d(unsigned lds_addr, const void* gaddr,
                                            unsigned tw, unsigned th,
                                            unsigned stride) {
  unsigned long long ga = (unsigned long long)(size_t)gaddr;
  v4u g0;
  g0[0] = 1u;                                                  // count=1
  g0[1] = lds_addr;                                            // lds byte addr
  g0[2] = (unsigned)ga;                                        // gaddr[31:0]
  g0[3] = (unsigned)((ga >> 32) & 0x01FFFFFFu) | (2u << 30);   // gaddr[56:32]|type=2
  v8i g1;
  g1[0] = 0x00010000;                            // wg_mask=0, data_size=1 (2B)
  g1[1] = (int)(tw << 16);                       // tensor_dim0 lo16
  g1[2] = (int)((Sn & 0xFFFFu) << 16);           // tdim0 hi | tensor_dim1 lo16
  g1[3] = (int)((tw & 0xFFFFu) << 16);           // tdim1 hi | tile_dim0
  g1[4] = (int)(th & 0xFFFFu);                   // tile_dim1, tile_dim2=0
  g1[5] = (int)stride;                           // tensor_dim0_stride lo32
  g1[6] = 0;
  g1[7] = 0;
  v4i z4 = {0, 0, 0, 0};
#if __clang_major__ >= 23
  v8i z8 = {0, 0, 0, 0, 0, 0, 0, 0};
  __builtin_amdgcn_tensor_load_to_lds(g0, g1, z4, z4, z8, 0);
#else
  __builtin_amdgcn_tensor_load_to_lds(g0, g1, z4, z4, 0);
#endif
}
#else
#define HAVE_TDM 0
#endif

// ---------------------------------------------------------------------------
// Kernel 0: weights fp32 -> bf16
// ---------------------------------------------------------------------------
__global__ __launch_bounds__(256) void convert_weights(
    const float* __restrict__ wqkv, const float* __restrict__ wo,
    bf16* __restrict__ wqkv_bf, bf16* __restrict__ wo_bf) {
  const int nq = 3 * Cn * Cn;
  int i = (blockIdx.x * 256 + threadIdx.x) * 4;
  if (i < nq) {
    v4f f = *(const v4f*)(wqkv + i);
    *(v2bf*)(wqkv_bf + i)     = pack_bf(f[0], f[1]);
    *(v2bf*)(wqkv_bf + i + 2) = pack_bf(f[2], f[3]);
  } else {
    int j = i - nq;
    v4f f = *(const v4f*)(wo + j);
    *(v2bf*)(wo_bf + j)     = pack_bf(f[0], f[1]);
    *(v2bf*)(wo_bf + j + 2) = pack_bf(f[2], f[3]);
  }
}

// ---------------------------------------------------------------------------
// Kernel 0b: x fp32 [b][c][s] -> xT bf16 [b][s][c]
// ---------------------------------------------------------------------------
__global__ __launch_bounds__(256) void transpose_x(
    const float* __restrict__ x, bf16* __restrict__ xt) {
  __shared__ float t[32][33];
  const int b = blockIdx.z;
  const int c0 = blockIdx.y * 32, s0 = blockIdx.x * 32;
  const int ts = threadIdx.x & 31, tc = threadIdx.x >> 5;
#pragma unroll
  for (int i = 0; i < 4; i++) {
    int c = tc + i * 8;
    t[c][ts] = x[((size_t)b * Cn + c0 + c) * Sn + s0 + ts];
  }
  __syncthreads();
  const int co = (threadIdx.x & 15) * 2, so = threadIdx.x >> 4;
#pragma unroll
  for (int i = 0; i < 2; i++) {
    int s = so + i * 16;
    *(v2bf*)(xt + ((size_t)b * Sn + s0 + s) * Cn + c0 + co) =
        pack_bf(t[co][s], t[co + 1][s]);
  }
}

// ---------------------------------------------------------------------------
// Kernel 1: QKV GEMM, 128x128 tile / 256 threads, K-step 64,
// async-LDS double-buffered staging; unroll 2 so buffer index is static.
// ---------------------------------------------------------------------------
__global__ __launch_bounds__(256) void qkv_gemm(
    const bf16* __restrict__ xt, const bf16* __restrict__ wqkv_bf,
    bf16* __restrict__ qws, bf16* __restrict__ kws, bf16* __restrict__ vtws) {
  __shared__ bf16 a_lds[2][128 * 64];   // 2 x 16KB
  __shared__ bf16 b_lds[2][128 * 64];
  const int tid = threadIdx.x;
  const int lane = tid & 31, wv = tid >> 5;
  const int wm = wv >> 1, wn = wv & 1;
  const int ln = lane & 15, lh = lane >> 4;
  const int n0 = blockIdx.x * 128;
  const int m0 = blockIdx.y * 128;
  const int b  = blockIdx.z;
  const int row = tid >> 1, half = tid & 1;

  const bf16* gA = wqkv_bf + (size_t)(m0 + row) * Cn + half * 32;
  const bf16* gB = xt + ((size_t)b * Sn + n0 + row) * Cn + half * 32;
  bf16* lA[2] = { a_lds[0] + row * 64 + half * 32, a_lds[1] + row * 64 + half * 32 };
  bf16* lB[2] = { b_lds[0] + row * 64 + half * 32, b_lds[1] + row * 64 + half * 32 };

  stage32_async(lA[0], gA);
  stage32_async(lB[0], gB);
  wait_async0();
  __syncthreads();

  v8f acc[2][4] = {};
#pragma unroll 2
  for (int ks = 0; ks < Cn / 64; ks++) {
    const int k0 = ks * 64;
    const int cur = ks & 1, nxt = cur ^ 1;
    if (k0 + 64 < Cn) {           // prefetch next tile while computing
      stage32_async(lA[nxt], gA + k0 + 64);
      stage32_async(lB[nxt], gB + k0 + 64);
    }
    const bf16* Ab = a_lds[cur];
    const bf16* Bb = b_lds[cur];
#pragma unroll
    for (int kk = 0; kk < 2; kk++) {
      v16bf af[2], bfr[4];
#pragma unroll
      for (int tm = 0; tm < 2; tm++)
        af[tm] = load_a_frag(Ab + (wm * 32 + tm * 16 + ln) * 64 + kk * 32, lh);
#pragma unroll
      for (int tn = 0; tn < 4; tn++)
        bfr[tn] = load_b_frag(Bb + (wn * 64 + tn * 16 + ln) * 64 + kk * 32, lh);
#pragma unroll
      for (int tm = 0; tm < 2; tm++)
#pragma unroll
        for (int tn = 0; tn < 4; tn++)
          acc[tm][tn] = wmma_bf16(af[tm], bfr[tn], acc[tm][tn]);
    }
    wait_async0();
    __syncthreads();
  }

  // wave-uniform q/k/v section (32-row wave tile inside one 64-ch section)
  const int obase = m0 + wm * 32;
  const int head  = __builtin_amdgcn_readfirstlane(obase / 192);
  const int secb  = obase - head * 192;
  const int sec   = __builtin_amdgcn_readfirstlane(secb >> 6);
  const int dd0   = secb & 63;
  const int hg    = b * NH + head;

  if (sec == 0) {          // Q, pre-scaled by 1/sqrt(d)*log2(e)
#pragma unroll
    for (int tm = 0; tm < 2; tm++) {
      int ddb = dd0 + tm * 16 + 8 * lh;
#pragma unroll
      for (int tn = 0; tn < 4; tn++) {
        int s = n0 + wn * 64 + tn * 16 + ln;
        bf16* dst = qws + ((size_t)hg * Sn + s) * Dn + ddb;
#pragma unroll
        for (int j = 0; j < 4; j++)
          *(v2bf*)(dst + 2 * j) =
              pack_bf(acc[tm][tn][2 * j] * QSCALE, acc[tm][tn][2 * j + 1] * QSCALE);
      }
    }
  } else if (sec == 1) {   // K
#pragma unroll
    for (int tm = 0; tm < 2; tm++) {
      int ddb = dd0 + tm * 16 + 8 * lh;
#pragma unroll
      for (int tn = 0; tn < 4; tn++) {
        int s = n0 + wn * 64 + tn * 16 + ln;
        bf16* dst = kws + ((size_t)hg * Sn + s) * Dn + ddb;
#pragma unroll
        for (int j = 0; j < 4; j++)
          *(v2bf*)(dst + 2 * j) = pack_bf(acc[tm][tn][2 * j], acc[tm][tn][2 * j + 1]);
      }
    }
  } else {                 // V, stored transposed [d][S]
#pragma unroll
    for (int tm = 0; tm < 2; tm++) {
      int ddb = dd0 + tm * 16 + 8 * lh;
#pragma unroll
      for (int tn = 0; tn < 4; tn++) {
        int s = n0 + wn * 64 + tn * 16 + ln;
#pragma unroll
        for (int r = 0; r < 8; r++)
          vtws[((size_t)hg * Dn + ddb + r) * Sn + s] = f2bf_trunc(acc[tm][tn][r]);
      }
    }
  }
}

// ---------------------------------------------------------------------------
// Kernel 2: flash attention (exp2 domain), double-buffered K/V chunks.
// Chunk staging via TDM (TENSOR_LOAD_TO_LDS, one wave) or async-LDS fallback.
// ---------------------------------------------------------------------------
__device__ __forceinline__ void attn_stage_chunk(
    int wv, int crow, int cpart, int hg, int kc,
    const bf16* kws, const bf16* vtws, bf16* kbuf, bf16* vbuf) {
#if HAVE_TDM
  if (wv == 0) {
    tdm_load_2d((unsigned)(size_t)kbuf,
                kws + ((size_t)hg * Sn + kc * 64) * Dn, 64, 64, Dn);
    tdm_load_2d((unsigned)(size_t)vbuf,
                vtws + (size_t)hg * Dn * Sn + kc * 64, 64, 64, Sn);
  }
#else
  const bf16* ksrc = kws + ((size_t)hg * Sn + kc * 64 + crow) * Dn + cpart * 16;
  const bf16* vsrc = vtws + ((size_t)hg * Dn + crow) * Sn + kc * 64 + cpart * 16;
  bf16* kd = kbuf + crow * 64 + cpart * 16;
  bf16* vd = vbuf + crow * 64 + cpart * 16;
  async_ld_b128((unsigned)(size_t)kd, (unsigned long long)(size_t)ksrc);
  async_ld_b128((unsigned)(size_t)(kd + 8), (unsigned long long)(size_t)(ksrc + 8));
  async_ld_b128((unsigned)(size_t)vd, (unsigned long long)(size_t)vsrc);
  async_ld_b128((unsigned)(size_t)(vd + 8), (unsigned long long)(size_t)(vsrc + 8));
#endif
}
__device__ __forceinline__ void attn_wait_chunk(int wv) {
#if HAVE_TDM
  if (wv == 0) __builtin_amdgcn_s_wait_tensorcnt(0);
#else
  wait_async0();
#endif
}

__global__ __launch_bounds__(256) void attn_kernel(
    const bf16* __restrict__ qws, const bf16* __restrict__ kws,
    const bf16* __restrict__ vtws, bf16* __restrict__ res) {
  __shared__ bf16 k_lds[2][64 * 64];     // [buf][key_local][dd]
  __shared__ bf16 v_lds[2][64 * 64];     // [buf][dd][key_local]
  __shared__ bf16 p_lds[8][16 * 64];     // per-wave P scratch [m][key_local]
  const int tid = threadIdx.x;
  const int lane = tid & 31, wv = tid >> 5;
  const int ln = lane & 15, lh = lane >> 4;
  const int hg = blockIdx.y;
  const int b = hg >> 3, head = hg & 7;
  const int q0 = blockIdx.x * 128 + wv * 16;
  const int crow = tid >> 2, cpart = tid & 3;

  const bf16* qrow = qws + ((size_t)hg * Sn + q0 + ln) * Dn;
  v16bf qf0 = load_a_frag(qrow, lh);
  v16bf qf1 = load_a_frag(qrow + 32, lh);

  v8f oacc[4] = {};
  float mrun[8], lrun[8];
#pragma unroll
  for (int r = 0; r < 8; r++) { mrun[r] = -3.0e38f; lrun[r] = 0.f; }

  attn_stage_chunk(wv, crow, cpart, hg, 0, kws, vtws, k_lds[0], v_lds[0]);
  attn_wait_chunk(wv);
  __syncthreads();

#pragma unroll 2
  for (int kc = 0; kc < Sn / 64; kc++) {
    const int cur = kc & 1, nxt = cur ^ 1;
    if (kc + 1 < Sn / 64)   // prefetch next chunk while computing this one
      attn_stage_chunk(wv, crow, cpart, hg, kc + 1, kws, vtws, k_lds[nxt], v_lds[nxt]);

    // S = Q K^T (16 x 64), already in log2 domain via QSCALE
    v8f sc[4];
#pragma unroll
    for (int ns = 0; ns < 4; ns++) {
      const bf16* col = k_lds[cur] + (ns * 16 + ln) * 64;
      v8f a = {};
      a = wmma_bf16(qf0, load_b_frag(col, lh), a);
      a = wmma_bf16(qf1, load_b_frag(col + 32, lh), a);
      sc[ns] = a;
    }

    float alpha[8];
#pragma unroll
    for (int r = 0; r < 8; r++) {
      float cm = fmaxf(fmaxf(sc[0][r], sc[1][r]), fmaxf(sc[2][r], sc[3][r]));
      cm = fmaxf(cm, __shfl_xor(cm, 1, 32));
      cm = fmaxf(cm, __shfl_xor(cm, 2, 32));
      cm = fmaxf(cm, __shfl_xor(cm, 4, 32));
      cm = fmaxf(cm, __shfl_xor(cm, 8, 32));
      float nm = fmaxf(mrun[r], cm);
      alpha[r] = fast_exp2(mrun[r] - nm);
      mrun[r] = nm;
    }
    bf16* pw = &p_lds[wv][0];
    float rsum[8];
#pragma unroll
    for (int r = 0; r < 8; r++) rsum[r] = 0.f;
#pragma unroll
    for (int ns = 0; ns < 4; ns++) {
#pragma unroll
      for (int r = 0; r < 8; r++) {
        float p = fast_exp2(sc[ns][r] - mrun[r]);
        rsum[r] += p;
        pw[(r + 8 * lh) * 64 + ns * 16 + ln] = f2bf_trunc(p);
      }
    }
#pragma unroll
    for (int r = 0; r < 8; r++) {
      float s = rsum[r];
      s += __shfl_xor(s, 1, 32);
      s += __shfl_xor(s, 2, 32);
      s += __shfl_xor(s, 4, 32);
      s += __shfl_xor(s, 8, 32);
      lrun[r] = alpha[r] * lrun[r] + s;
    }
#pragma unroll
    for (int nd = 0; nd < 4; nd++)
#pragma unroll
      for (int r = 0; r < 8; r++) oacc[nd][r] *= alpha[r];

    // O += P V  (same-wave LDS store->load is in-order)
    const bf16* prow = pw + ln * 64;
    v16bf pf0 = load_a_frag(prow, lh);
    v16bf pf1 = load_a_frag(prow + 32, lh);
#pragma unroll
    for (int nd = 0; nd < 4; nd++) {
      const bf16* col = v_lds[cur] + (nd * 16 + ln) * 64;
      oacc[nd] = wmma_bf16(pf0, load_b_frag(col, lh), oacc[nd]);
      oacc[nd] = wmma_bf16(pf1, load_b_frag(col + 32, lh), oacc[nd]);
    }
    attn_wait_chunk(wv);
    __syncthreads();
  }

#pragma unroll
  for (int r = 0; r < 8; r++) {
    int s = q0 + r + 8 * lh;
    float inv = 1.0f / lrun[r];
#pragma unroll
    for (int nd = 0; nd < 4; nd++) {
      int dd = nd * 16 + ln;
      res[((size_t)b * Sn + s) * Cn + head * Dn + dd] = f2bf_trunc(oacc[nd][r] * inv);
    }
  }
}

// ---------------------------------------------------------------------------
// Kernel 3: y = Wo @ result + bo + x (fp32 out), 128x128, async double-buffer
// ---------------------------------------------------------------------------
__global__ __launch_bounds__(256) void out_gemm(
    const float* __restrict__ x, const bf16* __restrict__ wo_bf,
    const bf16* __restrict__ res, const float* __restrict__ bo,
    float* __restrict__ y) {
  __shared__ bf16 a_lds[2][128 * 64];
  __shared__ bf16 b_lds[2][128 * 64];
  const int tid = threadIdx.x;
  const int lane = tid & 31, wv = tid >> 5;
  const int wm = wv >> 1, wn = wv & 1;
  const int ln = lane & 15, lh = lane >> 4;
  const int n0 = blockIdx.x * 128;
  const int m0 = blockIdx.y * 128;
  const int b  = blockIdx.z;
  const int row = tid >> 1, half = tid & 1;

  const bf16* gA = wo_bf + (size_t)(m0 + row) * Cn + half * 32;
  const bf16* gB = res + ((size_t)b * Sn + n0 + row) * Cn + half * 32;
  bf16* lA[2] = { a_lds[0] + row * 64 + half * 32, a_lds[1] + row * 64 + half * 32 };
  bf16* lB[2] = { b_lds[0] + row * 64 + half * 32, b_lds[1] + row * 64 + half * 32 };

  stage32_async(lA[0], gA);
  stage32_async(lB[0], gB);
  wait_async0();
  __syncthreads();

  v8f acc[2][4] = {};
#pragma unroll 2
  for (int ks = 0; ks < Cn / 64; ks++) {
    const int k0 = ks * 64;
    const int cur = ks & 1, nxt = cur ^ 1;
    if (k0 + 64 < Cn) {
      stage32_async(lA[nxt], gA + k0 + 64);
      stage32_async(lB[nxt], gB + k0 + 64);
    }
    const bf16* Ab = a_lds[cur];
    const bf16* Bb = b_lds[cur];
#pragma unroll
    for (int kk = 0; kk < 2; kk++) {
      v16bf af[2], bfr[4];
#pragma unroll
      for (int tm = 0; tm < 2; tm++)
        af[tm] = load_a_frag(Ab + (wm * 32 + tm * 16 + ln) * 64 + kk * 32, lh);
#pragma unroll
      for (int tn = 0; tn < 4; tn++)
        bfr[tn] = load_b_frag(Bb + (wn * 64 + tn * 16 + ln) * 64 + kk * 32, lh);
#pragma unroll
      for (int tm = 0; tm < 2; tm++)
#pragma unroll
        for (int tn = 0; tn < 4; tn++)
          acc[tm][tn] = wmma_bf16(af[tm], bfr[tn], acc[tm][tn]);
    }
    wait_async0();
    __syncthreads();
  }
#pragma unroll
  for (int tm = 0; tm < 2; tm++) {
#pragma unroll
    for (int r = 0; r < 8; r++) {
      int o = m0 + wm * 32 + tm * 16 + r + 8 * lh;
      float bias = bo[o];
#pragma unroll
      for (int tn = 0; tn < 4; tn++) {
        int s = n0 + wn * 64 + tn * 16 + ln;
        size_t idx = ((size_t)b * Cn + o) * Sn + s;
        y[idx] = acc[tm][tn][r] + bias + x[idx];
      }
    }
  }
}

// ---------------------------------------------------------------------------
extern "C" void kernel_launch(void* const* d_in, const int* in_sizes, int n_in,
                              void* d_out, int out_size, void* d_ws, size_t ws_size,
                              hipStream_t stream) {
  const float* x    = (const float*)d_in[0];
  const float* Wqkv = (const float*)d_in[1];
  const float* Wo   = (const float*)d_in[2];
  const float* bo   = (const float*)d_in[3];
  char* ws = (char*)d_ws;
  bf16* wqkv_bf = (bf16*)(ws + 0);
  bf16* wo_bf   = (bf16*)(ws + 1572864);
  bf16* xt      = (bf16*)(ws + 2097152);
  bf16* qws     = (bf16*)(ws + 10485760);
  bf16* kws     = (bf16*)(ws + 18874368);
  bf16* vtws    = (bf16*)(ws + 27262976);
  bf16* res     = (bf16*)(ws + 35651584);
  float* y = (float*)d_out;

  convert_weights<<<dim3(1024), dim3(256), 0, stream>>>(Wqkv, Wo, wqkv_bf, wo_bf);
  transpose_x<<<dim3(32, 16, 8), dim3(256), 0, stream>>>(x, xt);
  qkv_gemm<<<dim3(8, 12, 8), dim3(256), 0, stream>>>(xt, wqkv_bf, qws, kws, vtws);
  attn_kernel<<<dim3(8, 64), dim3(256), 0, stream>>>(qws, kws, vtws, res);
  out_gemm<<<dim3(8, 4, 8), dim3(256), 0, stream>>>(x, wo_bf, res, bo, y);
}
